// CustomAttention_30975304139057
// MI455X (gfx1250) — matrix operand order repaired and checked by reading, not compiled, not run on hardware
//
#include <hip/hip_runtime.h>
#include <hip/hip_bf16.h>

typedef __attribute__((ext_vector_type(16))) _Float16 v16h;
typedef __attribute__((ext_vector_type(8)))  _Float16 v8h;
typedef __attribute__((ext_vector_type(4)))  _Float16 v4h;
typedef __attribute__((ext_vector_type(2)))  __fp16   h2raw;   // cvt_pkrtz result type
typedef __attribute__((ext_vector_type(8)))  float    v8f;
typedef __attribute__((ext_vector_type(4)))  float    f32x4;

#define B_    4
#define SQ_   4096
#define SV_   4096
#define D_    64
#define QTILE 128          // q rows per block
#define KTILE 64           // keys per iteration
#define NJ    (KTILE / 16) // 4 key sub-tiles per chunk
#define NT    128          // threads per block (4 wave32)
#define WROWS 32           // q rows per wave (two 16-row WMMA tiles)

static __device__ __forceinline__ v8f wmma16x16x32(v16h a, v16h b, v8f c) {
  // D = A(16x32 f16) * B(32x16 f16) + C(16x16 f32)
  return __builtin_amdgcn_wmma_f32_16x16x32_f16(false, a, false, b, (short)0, c,
                                                false, false);
}

// A-matrix fragment (16x32 f16) from a row-major LDS tile.
// rowbase = &M[lane&15][k0]; half = lane>>4.
// Lanes 0-15 hold K {0..7,16..23}; lanes 16-31 hold K {8..15,24..31}.
static __device__ __forceinline__ v16h ld_afrag(const _Float16* rowbase, int half) {
  v8h lo = *(const v8h*)(rowbase + 8 * half);
  v8h hi = *(const v8h*)(rowbase + 8 * half + 16);
  v16h r;
#pragma unroll
  for (int i = 0; i < 8; ++i) { r[i] = lo[i]; r[i + 8] = hi[i]; }
  return r;
}

// B-matrix fragment (32x16 f16) where B[k][n] = M[n][k], M row-major in LDS.
// rowbase = &M[lane&15][k0]; half = lane>>4.
// Lanes 0-15 hold K {0..7,16..23} of column n = lane; lanes 16-31 hold K {8..15,24..31}.
static __device__ __forceinline__ v16h ld_bfrag(const _Float16* rowbase, int half) {
  v8h lo = *(const v8h*)(rowbase + 8 * half);
  v8h hi = *(const v8h*)(rowbase + 8 * half + 16);
  v16h r;
#pragma unroll
  for (int i = 0; i < 8; ++i) { r[i] = lo[i]; r[i + 8] = hi[i]; }
  return r;
}

__global__ __launch_bounds__(NT, 4)  // 4 waves/SIMD floor -> 256 VGPR budget
void fattn_qvv_kernel(const float* __restrict__ Q, const float* __restrict__ V,
                      float* __restrict__ O) {
  __shared__ alignas(16) _Float16 qs [QTILE][D_];    // 16 KB Q tile (f16)
  __shared__ alignas(16) _Float16 vs [KTILE][D_];    //  8 KB V chunk row-major
  __shared__ alignas(16) _Float16 vsT[D_][KTILE];    //  8 KB V chunk transposed

  const int b    = blockIdx.y;
  const int q0   = blockIdx.x * QTILE;
  const int tid  = threadIdx.x;
  const int lane = tid & 31;
  const int wave = tid >> 5;
  const int mrow = lane & 15;  // A-frag row / B-frag column owned by this lane
  const int half = lane >> 4;

  const float* Qb = Q + (size_t)b * SQ_ * D_;
  const float* Vb = V + (size_t)b * SV_ * D_;
  float*       Ob = O + (size_t)b * SQ_ * D_;

  // Stage Q tile as f16 (stays resident in LDS; fragments reloaded at use).
  for (int i = tid; i < QTILE * D_ / 4; i += NT) {
    const int row = i >> 4;  // 16 float4 per 64-wide row
    const int c4  = i & 15;
    f32x4 v = ((const f32x4*)(Qb + (size_t)(q0 + row) * D_))[c4];
    _Float16* dst = &qs[row][c4 * 4];
    dst[0] = (_Float16)v[0]; dst[1] = (_Float16)v[1];
    dst[2] = (_Float16)v[2]; dst[3] = (_Float16)v[3];
  }
  __syncthreads();

  // Ones B-fragment: osum += P x ones gives per-row softmax denominators
  // in the same C-layout as the O accumulators.
  v16h ones;
#pragma unroll
  for (int i = 0; i < 16; ++i) ones[i] = (_Float16)1.0f;

  v8f oacc[2][4] = {{{}, {}, {}, {}}, {{}, {}, {}, {}}};
  v8f osum[2]    = {{}, {}};
  float m[2]     = {-INFINITY, -INFINITY};

  for (int kv0 = 0; kv0 < SV_; kv0 += KTILE) {
    __syncthreads();  // previous chunk fully consumed before overwrite
    // Stage V chunk: two rows per step so the transpose becomes packed
    // b32 stores (v_cvt_pk_rtz_f16_f32 pairs rows rp/rp+1).
    for (int i = tid; i < KTILE * D_ / 8; i += NT) {
      const int rp = (i >> 4) << 1;  // even row
      const int c4 = i & 15;
      f32x4 v0 = ((const f32x4*)(Vb + (size_t)(kv0 + rp)     * D_))[c4];
      f32x4 v1 = ((const f32x4*)(Vb + (size_t)(kv0 + rp + 1) * D_))[c4];
      h2raw p0 = __builtin_amdgcn_cvt_pkrtz(v0[0], v1[0]);
      h2raw p1 = __builtin_amdgcn_cvt_pkrtz(v0[1], v1[1]);
      h2raw p2 = __builtin_amdgcn_cvt_pkrtz(v0[2], v1[2]);
      h2raw p3 = __builtin_amdgcn_cvt_pkrtz(v0[3], v1[3]);
      v4h r0 = {(_Float16)p0[0], (_Float16)p1[0], (_Float16)p2[0], (_Float16)p3[0]};
      v4h r1 = {(_Float16)p0[1], (_Float16)p1[1], (_Float16)p2[1], (_Float16)p3[1]};
      *(v4h*)&vs[rp][c4 * 4]     = r0;
      *(v4h*)&vs[rp + 1][c4 * 4] = r1;
      *(h2raw*)&vsT[c4 * 4 + 0][rp] = p0;
      *(h2raw*)&vsT[c4 * 4 + 1][rp] = p1;
      *(h2raw*)&vsT[c4 * 4 + 2][rp] = p2;
      *(h2raw*)&vsT[c4 * 4 + 3][rp] = p3;
    }
    __syncthreads();

#pragma unroll
    for (int t = 0; t < 2; ++t) {
      // Q B-fragments reloaded from LDS at use (cheaper than scratch spills).
      const _Float16* qrow = &qs[wave * WROWS + t * 16 + mrow][0];
      const v16h qb0 = ld_bfrag(qrow,      half);
      const v16h qb1 = ld_bfrag(qrow + 32, half);

      // GEMM1: Sᵀ(keys x q). C-layout: row = key (r + 8*half), col = q-row.
      v8f s[NJ];
#pragma unroll
      for (int j = 0; j < NJ; ++j) {
        const _Float16* vrow = &vs[j * 16 + mrow][0];
        v8f acc = {};
        acc  = wmma16x16x32(ld_afrag(vrow,      half), qb0, acc);
        s[j] = wmma16x16x32(ld_afrag(vrow + 32, half), qb1, acc);
      }

      // Each lane owns one q-row (= lane&15): per-lane max over its 32 keys,
      // then a single xor-16 shuffle merges the two key halves.
      float rm = -INFINITY;
#pragma unroll
      for (int j = 0; j < NJ; ++j)
#pragma unroll
        for (int r = 0; r < 8; ++r) rm = fmaxf(rm, s[j][r]);
      rm = fmaxf(rm, __shfl_xor(rm, 16));
      const float mn    = fmaxf(m[t], rm);
      const float alpha = __expf(m[t] - mn);
      m[t] = mn;

      // P = exp(Sᵀ - m): the Sᵀ C-layout *is* the A-frag layout of P.
      // pa0 <- key sub-tiles j=0,1 ; pa1 <- j=2,3.
      v16h pa0, pa1;
#pragma unroll
      for (int j = 0; j < NJ; ++j) {
        const int base = (j & 1) * 8;
#pragma unroll
        for (int i = 0; i < 4; ++i) {
          h2raw pk = __builtin_amdgcn_cvt_pkrtz(__expf(s[j][2 * i]     - mn),
                                                __expf(s[j][2 * i + 1] - mn));
          if (j < 2) {
            pa0[base + 2 * i]     = (_Float16)pk[0];
            pa0[base + 2 * i + 1] = (_Float16)pk[1];
          } else {
            pa1[base + 2 * i]     = (_Float16)pk[0];
            pa1[base + 2 * i + 1] = (_Float16)pk[1];
          }
        }
      }

      // Rescale accumulators: alpha lives per-lane (q-row = lane&15);
      // O/osum rows are (r + 8*half) -> broadcast from lane (r + 8*half).
#pragma unroll
      for (int r = 0; r < 8; ++r) {
        const float aR = __shfl(alpha, r + 8 * half);
        osum[t][r] *= aR;
#pragma unroll
        for (int n = 0; n < 4; ++n) oacc[t][n][r] *= aR;
      }

      // Row sums via ones column, then GEMM2: O += P x V.
      osum[t] = wmma16x16x32(pa0, ones, osum[t]);
      osum[t] = wmma16x16x32(pa1, ones, osum[t]);
#pragma unroll
      for (int n = 0; n < 4; ++n) {
        const _Float16* vtrow = &vsT[n * 16 + mrow][0];
        oacc[t][n] = wmma16x16x32(pa0, ld_bfrag(vtrow,      half), oacc[t][n]);
        oacc[t][n] = wmma16x16x32(pa1, ld_bfrag(vtrow + 32, half), oacc[t][n]);
      }

      // Keep the two q-tiles' live ranges separate (avoid register blowup
      // from cross-tile software pipelining; waves provide the overlap).
      asm volatile("" ::: "memory");
    }
  }

  // Normalize (osum already in O's C-layout) and write fp32 output.
#pragma unroll
  for (int t = 0; t < 2; ++t) {
#pragma unroll
    for (int r = 0; r < 8; ++r) {
      const float inv = 1.0f / osum[t][r];
      const int row   = q0 + wave * WROWS + t * 16 + r + 8 * half;
      float* orow = Ob + (size_t)row * D_;
#pragma unroll
      for (int n = 0; n < 4; ++n) orow[n * 16 + mrow] = oacc[t][n][r] * inv;
    }
  }
}

extern "C" void kernel_launch(void* const* d_in, const int* in_sizes, int n_in,
                              void* d_out, int out_size, void* d_ws, size_t ws_size,
                              hipStream_t stream) {
  const float* Q = (const float*)d_in[0];
  const float* V = (const float*)d_in[1];
  float* O = (float*)d_out;
  dim3 grid(SQ_ / QTILE, B_, 1);
  dim3 block(NT, 1, 1);
  hipLaunchKernelGGL(fattn_qvv_kernel, grid, block, 0, stream, Q, V, O);
}
